// WidthCharNameTransformer_63754494542649
// MI455X (gfx1250) — compile-verified, matrix operand order checked by reading, not run on hardware
//
#include <hip/hip_runtime.h>
#include <hip/hip_bf16.h>
#include <stdint.h>

// ---------------------------------------------------------------------------
// MI455X (gfx1250): all dense GEMMs on v_wmma_f32_16x16x32_f16.
//  - weights pre-repacked into WMMA B-fragment layout -> 2x global_load_b128/lane
//  - activations in LDS with addrspace(3) pointers -> 2x ds_load_b128 per A frag
//  - A fragments register-cached and reused across a wave's N-tiles
// wave32, 8 waves / 256-thread workgroup.
// ---------------------------------------------------------------------------

#define AS1 __attribute__((address_space(1)))
#define AS3 __attribute__((address_space(3)))

typedef __attribute__((ext_vector_type(16))) _Float16 v16h;
typedef __attribute__((ext_vector_type(8)))  float    v8f;
typedef __attribute__((ext_vector_type(4)))  uint32_t u32x4;

union ABReg  { v16h v; u32x4 q[2]; _Float16 h[16]; };
union Row32h { u32x4 q[4]; _Float16 h[32]; };

template <typename T>
__device__ __forceinline__ AS1 T* as_global(T* p) {
  return (AS1 T*)(uintptr_t)p;                 // global addr == generic addr
}
template <typename T>
__device__ __forceinline__ AS3 T* as_lds(T* p) {
  return (AS3 T*)(uint32_t)(uintptr_t)p;       // generic-LDS low 32b = LDS offset
}

__device__ __forceinline__ unsigned laneid() { return threadIdx.x & 31u; }
__device__ __forceinline__ int      waveid() { return threadIdx.x >> 5; }

__device__ __forceinline__ float wsum(float v) {
#pragma unroll
  for (int o = 16; o > 0; o >>= 1) v += __shfl_xor(v, o, 32);
  return v;
}
__device__ __forceinline__ float wmaxf(float v) {
#pragma unroll
  for (int o = 16; o > 0; o >>= 1) v = fmaxf(v, __shfl_xor(v, o, 32));
  return v;
}

// ---- A fragment from LDS (16-bit A 16x32 ISA layout) -----------------------
// lanes 0-15: M=lane, K={0..7,16..23}; lanes 16-31: M=lane-16, K=+8
// -> two ds_load_b128 per fragment.
__device__ __forceinline__ v16h load_A_lds(const AS3 _Float16* a, int row0,
                                           int k0, int ld) {
  const unsigned l = laneid();
  const int m  = row0 + (int)(l & 15u);
  const int kh = (int)((l >> 4) << 3);
  const AS3 u32x4* p = (const AS3 u32x4*)(a + m * ld + k0 + kh);
  ABReg r;
  r.q[0] = p[0];      // K 0..7   (+kh)
  r.q[1] = p[2];      // K 16..23 (+kh)
  return r.v;
}

// ---- B fragment from repacked weights: blob of 32 lanes x 16 halves --------
// -> two global_load_b128 per fragment, fully coalesced (1KB/wave contiguous).
__device__ __forceinline__ v16h load_B_frag(const AS1 _Float16* B, int blk) {
  const AS1 u32x4* p =
      (const AS1 u32x4*)(B + ((size_t)blk * 32 + laneid()) * 16);
  ABReg r;
  r.q[0] = p[0];
  r.q[1] = p[1];
  return r.v;
}

// ---- block GEMM: C[MT*16 x N] = A[MT*16 x KS*32] @ B ------------------------
// N = NTPW*128 (NTPW tiles per wave). MODE 0: f16 store, 1: relu->f16, 2: f32+resid
template <int MT, int NTPW, int KS, int MODE>
__device__ void gemm_wmma(const AS3 _Float16* A, const AS1 _Float16* B,
                          AS3 _Float16* outh, AS3 float* outf,
                          const AS3 float* resid) {
  constexpr int N  = NTPW * 8 * 16;
  constexpr int K  = KS * 32;
  constexpr int CH = (KS < 8) ? KS : 8;        // K-chunk: A frags cached in regs
  const int wv = waveid();
  const unsigned l = laneid();
  for (int mt = 0; mt < MT; ++mt) {
    v8f acc[NTPW];
#pragma unroll
    for (int i = 0; i < NTPW; ++i) { v8f z = {}; acc[i] = z; }
    for (int kc = 0; kc < KS; kc += CH) {
      v16h af[CH];
#pragma unroll
      for (int s = 0; s < CH; ++s)
        af[s] = load_A_lds(A, mt * 16, (kc + s) * 32, K);
#pragma unroll
      for (int i = 0; i < NTPW; ++i) {
        const int nt = wv + 8 * i;
#pragma unroll
        for (int s = 0; s < CH; ++s) {
          v16h b = load_B_frag(B, nt * KS + kc + s);
          acc[i] = __builtin_amdgcn_wmma_f32_16x16x32_f16(
              false, af[s], false, b, (short)0, acc[i], false, false);
        }
      }
    }
    // D layout: lanes 0-15 -> N=lane, M=r; lanes 16-31 -> N=lane-16, M=r+8
    const int nb = (int)(l & 15u);
    const int mb = mt * 16 + (int)((l >> 4) << 3);
#pragma unroll
    for (int i = 0; i < NTPW; ++i) {
      const int n = (wv + 8 * i) * 16 + nb;
#pragma unroll
      for (int r = 0; r < 8; ++r) {
        const int row = mb + r;
        const float v = acc[i][r];
        if (MODE == 0)      outh[row * N + n] = (_Float16)v;
        else if (MODE == 1) outh[row * N + n] = (_Float16)(v > 0.f ? v : 0.f);
        else                outf[row * N + n] = v + resid[row * N + n];
      }
    }
  }
}

// ---- attention: one wave per head (H=8, hd=32) ------------------------------
// qkv rows x 768 f16 in LDS (Q 0..255 | K 256..511 | V 512..767)
__device__ void attention(const AS3 _Float16* qkv, unsigned keymask, int nq,
                          int nkeys, AS3 float* sbuf, AS3 float* arow) {
  const int h = waveid();
  const unsigned l = laneid();
  const int kr = (l < (unsigned)nkeys) ? (int)l : 0;
  const bool kvalid = (l < (unsigned)nkeys) && ((keymask >> l) & 1u);
  Row32h kb;
  {
    const AS3 u32x4* kp = (const AS3 u32x4*)(qkv + kr * 768 + 256 + h * 32);
#pragma unroll
    for (int j = 0; j < 4; ++j) kb.q[j] = kp[j];
  }
  const float scale = 0.17677669529663687f;    // 1/sqrt(32)
  for (int q = 0; q < nq; ++q) {
    Row32h qb;
    const AS3 u32x4* qp = (const AS3 u32x4*)(qkv + q * 768 + h * 32);
#pragma unroll
    for (int j = 0; j < 4; ++j) qb.q[j] = qp[j];
    float s = 0.f;                             // lane = key index
#pragma unroll
    for (int d = 0; d < 32; ++d) s += (float)qb.h[d] * (float)kb.h[d];
    s = kvalid ? s * scale : -1e9f;            // pad keys -> -1e9 (as reference)
    const float m = wmaxf(s);
    const float p = __expf(s - m);
    const float sum = wsum(p);
    arow[h * 32 + (int)l] = p / sum;           // same-wave LDS: in-order
    float o = 0.f;                             // lane = head dim
    for (int k = 0; k < nkeys; ++k)
      o += arow[h * 32 + k] * (float)qkv[k * 768 + 512 + h * 32 + (int)l];
    sbuf[q * 256 + h * 32 + (int)l] = o;
  }
}

// ---- layernorm over 256-wide rows: one wave per row -------------------------
template <bool WRITE_H>
__device__ void layernorm(const AS3 float* in, int rows, AS3 float* of,
                          AS3 _Float16* oh) {
  const int l = (int)laneid();
  for (int r = waveid(); r < rows; r += 8) {
    const AS3 float* x = in + r * 256;
    float s = 0.f;
    for (int d = l; d < 256; d += 32) s += x[d];
    const float mean = wsum(s) * (1.f / 256.f);
    float vv = 0.f;
    for (int d = l; d < 256; d += 32) { float t = x[d] - mean; vv += t * t; }
    const float inv = rsqrtf(wsum(vv) * (1.f / 256.f) + 1e-5f);
    for (int d = l; d < 256; d += 32) {
      const float y = (x[d] - mean) * inv;
      of[r * 256 + d] = y;
      if (WRITE_H) oh[r * 256 + d] = (_Float16)y;
    }
  }
}

// ---------------------------------------------------------------------------
// Word-level block: one workgroup per word. L=24 padded to 32 rows.
// LDS: xf 32K | xh 16K | sbuf 32K | qh 64K | arow 1K | chs 128B  = ~145KB
// ---------------------------------------------------------------------------
__global__ __launch_bounds__(256) void word_kernel(
    const int* __restrict__ inputs_p, const int* __restrict__ concept_id_p,
    const int* __restrict__ word_pos_p, const float* __restrict__ emb_p,
    const _Float16* __restrict__ wqkv_p, const _Float16* __restrict__ wo_p,
    const _Float16* __restrict__ wff1_p, const _Float16* __restrict__ wff2_p,
    float* __restrict__ padded_p) {
  extern __shared__ char smem[];
  AS3 float*    xf   = as_lds((float*)smem);                       // 32x256
  AS3 _Float16* xh   = as_lds((_Float16*)(smem + 32768));          // 32x256
  AS3 float*    sbuf = as_lds((float*)(smem + 49152));             // 32x256
  AS3 _Float16* qh   = as_lds((_Float16*)(smem + 81920));          // 32x1024
  AS3 float*    arow = as_lds((float*)(smem + 147456));            // 8x32
  AS3 int*      chs  = as_lds((int*)(smem + 148480));              // 32

  const AS1 int*      inputs = as_global(inputs_p);
  const AS1 float*    emb    = as_global(emb_p);
  const AS1 _Float16* Bqkv   = as_global(wqkv_p);
  const AS1 _Float16* Bo     = as_global(wo_p);
  const AS1 _Float16* Bff1   = as_global(wff1_p);
  const AS1 _Float16* Bff2   = as_global(wff2_p);

  const int w = blockIdx.x, tid = threadIdx.x;
  if (tid < 32) chs[tid] = (tid < 24) ? inputs[w * 24 + tid] : 0;
  __syncthreads();

  // embedding gather (pad chars use emb row 0 == we_emb[inputs])
  for (int c = 0; c < 32; ++c) {
    const float v = (c < 24) ? emb[chs[c] * 256 + tid] : 0.f;
    xf[c * 256 + tid] = v;
    xh[c * 256 + tid] = (_Float16)v;
  }
  __syncthreads();

  // QKV: [32x256] @ [256x768]
  gemm_wmma<2, 6, 8, 0>(xh, Bqkv, qh, sbuf, xf);
  __syncthreads();

  unsigned keymask = 0; int nvalid = 0;
  for (int c = 0; c < 24; ++c) if (chs[c] != 0) { keymask |= 1u << c; ++nvalid; }
  attention(qh, keymask, 24, 24, sbuf, arow);
  __syncthreads();

  for (int c = 0; c < 24; ++c)                 // attn -> f16 (rows 24.. stay 0)
    xh[c * 256 + tid] = (_Float16)sbuf[c * 256 + tid];
  __syncthreads();

  // O-proj + residual: sbuf = x + attn @ wo
  gemm_wmma<2, 2, 8, 2>(xh, Bo, qh, sbuf, xf);
  __syncthreads();
  layernorm<true>(sbuf, 32, xf, xh);           // h -> xf(f32), xh(f16)
  __syncthreads();

  // FF1 + relu: qh = relu(h @ w1) [32x1024] f16
  gemm_wmma<2, 8, 8, 1>(xh, Bff1, qh, sbuf, xf);
  __syncthreads();
  // FF2 + residual: sbuf = h + ff @ w2
  gemm_wmma<2, 2, 32, 2>(qh, Bff2, xh, sbuf, xf);
  __syncthreads();
  layernorm<false>(sbuf, 32, xf, xh);
  __syncthreads();

  // masked mean over valid chars + fused scatter
  float s = 0.f;
  for (int c = 0; c < 24; ++c)
    if ((keymask >> c) & 1u) s += xf[c * 256 + tid];
  AS1 float* padded = as_global(padded_p);
  padded[((size_t)concept_id_p[w] * 16 + word_pos_p[w]) * 256 + tid] =
      s / (float)nvalid;
}

// ---------------------------------------------------------------------------
// Name-level block + pooling: one workgroup per concept, 16 rows (1 M-tile).
// key_pad=None in reference -> all 16 keys attend (zero rows included).
// ---------------------------------------------------------------------------
__global__ __launch_bounds__(256) void name_kernel(
    const float* __restrict__ padded_p, const _Float16* __restrict__ wqkv_p,
    const _Float16* __restrict__ wo_p, const _Float16* __restrict__ wff1_p,
    const _Float16* __restrict__ wff2_p, float* __restrict__ out_p) {
  extern __shared__ char smem[];
  AS3 float*    xf    = as_lds((float*)smem);                      // 16x256
  AS3 _Float16* xh    = as_lds((_Float16*)(smem + 16384));         // 16x256
  AS3 float*    sbuf  = as_lds((float*)(smem + 24576));            // 16x256
  AS3 _Float16* qh    = as_lds((_Float16*)(smem + 40960));         // 16x1024
  AS3 float*    arow  = as_lds((float*)(smem + 73728));            // 8x32
  AS3 unsigned* rowfl = as_lds((unsigned*)(smem + 74752));         // 16

  const AS1 float*    padded = as_global(padded_p);
  const AS1 _Float16* Bqkv   = as_global(wqkv_p);
  const AS1 _Float16* Bo     = as_global(wo_p);
  const AS1 _Float16* Bff1   = as_global(wff1_p);
  const AS1 _Float16* Bff2   = as_global(wff2_p);

  const int ci = blockIdx.x, tid = threadIdx.x;
  for (int c = 0; c < 16; ++c) {
    const float v = padded[(size_t)ci * 4096 + c * 256 + tid];
    xf[c * 256 + tid] = v;
    xh[c * 256 + tid] = (_Float16)v;
  }
  __syncthreads();
  if (tid < 16) {                              // non-pad row detection
    unsigned f = 0;
    for (int d = 0; d < 256; ++d) f |= (xf[tid * 256 + d] != 0.f) ? 1u : 0u;
    rowfl[tid] = f;
  }
  __syncthreads();
  unsigned rnz = 0;
  for (int c = 0; c < 16; ++c) rnz |= rowfl[c] << c;

  gemm_wmma<1, 6, 8, 0>(xh, Bqkv, qh, sbuf, xf);
  __syncthreads();
  attention(qh, 0xFFFFu, 16, 16, sbuf, arow);  // no key mask
  __syncthreads();
  for (int c = 0; c < 16; ++c)
    xh[c * 256 + tid] = (_Float16)sbuf[c * 256 + tid];
  __syncthreads();
  gemm_wmma<1, 2, 8, 2>(xh, Bo, qh, sbuf, xf);
  __syncthreads();
  layernorm<true>(sbuf, 16, xf, xh);
  __syncthreads();
  gemm_wmma<1, 8, 8, 1>(xh, Bff1, qh, sbuf, xf);
  __syncthreads();
  gemm_wmma<1, 2, 32, 2>(qh, Bff2, xh, sbuf, xf);
  __syncthreads();
  layernorm<false>(sbuf, 16, xf, xh);
  __syncthreads();

  // zero pad rows, per-feature nonzero count, pooled mean -> d_out
  float s = 0.f, cnt = 0.f;
  for (int c = 0; c < 16; ++c) {
    const float v = ((rnz >> c) & 1u) ? xf[c * 256 + tid] : 0.f;
    s += v;
    cnt += (v != 0.f) ? 1.f : 0.f;
  }
  AS1 float* out = as_global(out_p);
  out[(size_t)ci * 256 + tid] = s / cnt;
}

// ---- weight repack: f32 [K x N] -> f16 WMMA B-fragment blobs ---------------
// dst[((nt*(K/32)+ks)*32 + lane)*16 + j] = (f16) src[k*N + n]
//   n = nt*16 + (lane&15); kh = (lane>>4)*8; jp=j>>1
//   k = ks*32 + (jp<4 ? 2*jp : 2*jp+8) + kh + (j&1)
__global__ void repack_B_kernel(const float* __restrict__ src,
                                _Float16* __restrict__ dst, int K, int N) {
  const int idx = blockIdx.x * blockDim.x + threadIdx.x;
  if (idx >= K * N) return;
  const int j    = idx & 15;
  const int lane = (idx >> 4) & 31;
  const int blk  = idx >> 9;
  const int ksteps = K >> 5;
  const int nt = blk / ksteps, ks = blk % ksteps;
  const int n  = nt * 16 + (lane & 15);
  const int kh = (lane >> 4) << 3;
  const int jp = j >> 1;
  const int k  = ks * 32 + (jp < 4 ? 2 * jp : 2 * jp + 8) + kh + (j & 1);
  dst[idx] = (_Float16)src[k * N + n];
}

__global__ void zero_f32_kernel(float* __restrict__ p, int n) {
  const int i = blockIdx.x * blockDim.x + threadIdx.x;
  if (i < n) p[i] = 0.f;
}

// ---------------------------------------------------------------------------
extern "C" void kernel_launch(void* const* d_in, const int* in_sizes, int n_in,
                              void* d_out, int out_size, void* d_ws,
                              size_t ws_size, hipStream_t stream) {
  (void)in_sizes; (void)n_in; (void)out_size; (void)ws_size;
  const int*   inputs     = (const int*)d_in[0];
  const int*   concept_id = (const int*)d_in[1];
  const int*   word_pos   = (const int*)d_in[2];
  const float* we_emb     = (const float*)d_in[5];

  char* ws = (char*)d_ws;
  size_t off = 0;
  auto alloc = [&](size_t bytes) {
    char* p = ws + off; off += (bytes + 255) & ~size_t(255); return p;
  };
  float*    padded = (float*)alloc((size_t)1024 * 16 * 256 * 4);   // 16 MB
  _Float16* wB[8];
  const int KN[8][2] = {{256, 768}, {256, 256}, {256, 1024}, {1024, 256},
                        {256, 768}, {256, 256}, {256, 1024}, {1024, 256}};
  for (int i = 0; i < 8; ++i)
    wB[i] = (_Float16*)alloc((size_t)KN[i][0] * KN[i][1] * 2);

  for (int i = 0; i < 8; ++i) {
    const int n = KN[i][0] * KN[i][1];
    repack_B_kernel<<<(n + 255) / 256, 256, 0, stream>>>(
        (const float*)d_in[6 + i], wB[i], KN[i][0], KN[i][1]);
  }
  {
    const int n = 1024 * 16 * 256;
    zero_f32_kernel<<<(n + 255) / 256, 256, 0, stream>>>(padded, n);
  }

  const size_t word_smem = 148608;   // xf+xh+sbuf+qh+arow+chs
  word_kernel<<<8192, 256, word_smem, stream>>>(
      inputs, concept_id, word_pos, we_emb, wB[0], wB[1], wB[2], wB[3], padded);

  const size_t name_smem = 74816;    // xf+xh+sbuf+qh+arow+rowflags
  name_kernel<<<1024, 256, name_smem, stream>>>(
      padded, wB[4], wB[5], wB[6], wB[7], (float*)d_out);
}
